// GraphEncoder_75857712382359
// MI455X (gfx1250) — compile-verified, matrix operand order checked by reading, not run on hardware
//
#include <hip/hip_runtime.h>
#include <hip/hip_bf16.h>
#include <hip/hip_fp16.h>

typedef __attribute__((ext_vector_type(16))) _Float16 v16h;
typedef __attribute__((ext_vector_type(8)))  _Float16 v8h;
typedef __attribute__((ext_vector_type(8)))  float    v8f;

#define HH   4
#define IND  16
#define HC1  256
#define CC1  64
#define HC2  128
#define CC2  32
#define NEG  0.2f

// ---------- helpers ----------
static __device__ __forceinline__ unsigned flipf(float f) {
  unsigned u = __float_as_uint(f);
  return (u >> 31) ? ~u : (u | 0x80000000u);
}
static __device__ __forceinline__ float unflipf(unsigned u) {
  return __uint_as_float((u >> 31) ? (u & 0x7fffffffu) : ~u);
}

// Load one WMMA f16 fragment (A or B) from a row-major f16 matrix whose
// "line" stride is stride_h halves. Per ISA 16-bit layout: lane l handles
// line (l&15); halves 0..7 = K[(l>>4)*8 .. +7], halves 8..15 = K[16+(l>>4)*8 .. +7].
static __device__ __forceinline__ v16h frag_load(const _Float16* p) {
  v8h lo = *(const v8h*)p;
  v8h hi = *(const v8h*)(p + 16);
  v16h r;
#pragma unroll
  for (int i = 0; i < 8; ++i) { r[i] = lo[i]; r[i + 8] = hi[i]; }
  return r;
}

// ---------- conversion / precompute ----------
__global__ void k_cvt_x(const float* __restrict__ x, _Float16* __restrict__ xpad, int n) {
  int idx = blockIdx.x * blockDim.x + threadIdx.x;
  if (idx >= n * 32) return;
  int r = idx >> 5, k = idx & 31;
  xpad[idx] = (k < IND) ? (_Float16)x[r * IND + k] : (_Float16)0.0f;
}
__global__ void k_cvt_w1(const float* __restrict__ W1, _Float16* __restrict__ w1t) {
  int idx = blockIdx.x * blockDim.x + threadIdx.x;
  if (idx >= HC1 * 32) return;
  int col = idx >> 5, k = idx & 31;          // w1t[col][k], K padded 16->32
  w1t[idx] = (k < IND) ? (_Float16)W1[k * HC1 + col] : (_Float16)0.0f;
}
__global__ void k_cvt_w2(const float* __restrict__ W2, _Float16* __restrict__ w2t) {
  int idx = blockIdx.x * blockDim.x + threadIdx.x;
  if (idx >= HC2 * HC1) return;
  int col = idx >> 8, k = idx & 255;         // w2t[col][k]
  w2t[idx] = (_Float16)W2[k * HC2 + col];
}
// ve[h*4+d] = sum_c We[d, h*C+c] * ae[h,c]   (collapses edge embedding to 4x4)
__global__ void k_ve(const float* __restrict__ We1, const float* __restrict__ ae1,
                     const float* __restrict__ We2, const float* __restrict__ ae2,
                     float* __restrict__ ve) {
  int t = threadIdx.x;
  if (t < 16) {
    int h = t >> 2, d = t & 3; float s = 0.f;
    for (int c = 0; c < CC1; ++c) s += We1[d * HC1 + h * CC1 + c] * ae1[h * CC1 + c];
    ve[t] = s;
  } else if (t < 32) {
    int q = t - 16; int h = q >> 2, d = q & 3; float s = 0.f;
    for (int c = 0; c < CC2; ++c) s += We2[d * HC2 + h * CC2 + c] * ae2[h * CC2 + c];
    ve[16 + q] = s;
  }
}

// ---------- self-loop edge_attr = mean of incoming ----------
__global__ void k_cntsum(const int* __restrict__ dst, const float* __restrict__ ea,
                         float* __restrict__ cnt, float* __restrict__ lea, int e) {
  int i = blockIdx.x * blockDim.x + threadIdx.x;
  if (i >= e) return;
  int d = dst[i];
  atomicAdd(&cnt[d], 1.0f);
#pragma unroll
  for (int j = 0; j < 4; ++j) atomicAdd(&lea[d * 4 + j], ea[(size_t)i * 4 + j]);
}
__global__ void k_loopdiv(float* __restrict__ lea, const float* __restrict__ cnt, int n) {
  int idx = blockIdx.x * blockDim.x + threadIdx.x;
  if (idx >= n * 4) return;
  float c = cnt[idx >> 2]; c = (c < 1.0f) ? 1.0f : c;
  lea[idx] /= c;
}

// ---------- WMMA GEMMs ----------
// xh1[N,256] = xpad[N,32(f16)] @ w1t^T  ; 16 rows per block, 8 waves, 2 tiles/wave
__global__ void k_gemm1(const _Float16* __restrict__ xpad, const _Float16* __restrict__ w1t,
                        float* __restrict__ xh1, int n) {
  int wave = threadIdx.x >> 5, lane = threadIdx.x & 31;
  int row0 = blockIdx.x * 16;
  if (row0 + 16 > n) return;
  const _Float16* pa = xpad + (size_t)(row0 + (lane & 15)) * 32 + ((lane >> 4) << 3);
  v16h a = frag_load(pa);
#pragma unroll
  for (int t = 0; t < 2; ++t) {
    int col0 = (wave * 2 + t) * 16;
    const _Float16* pb = w1t + (size_t)(col0 + (lane & 15)) * 32 + ((lane >> 4) << 3);
    v16h b = frag_load(pb);
    v8f c = {};
    c = __builtin_amdgcn_wmma_f32_16x16x32_f16(false, a, false, b, (short)0, c, false, false);
    int rb = row0 + ((lane >> 4) << 3);
    int col = col0 + (lane & 15);
#pragma unroll
    for (int j = 0; j < 8; ++j) xh1[(size_t)(rb + j) * HC1 + col] = c[j];
  }
}
// xh2[N,128] = h1[N,256(f16)] @ w2t^T ; 8 waves = 8 col tiles, K-loop of 8 WMMAs
__global__ void k_gemm2(const _Float16* __restrict__ h1, const _Float16* __restrict__ w2t,
                        float* __restrict__ xh2, int n) {
  int wave = threadIdx.x >> 5, lane = threadIdx.x & 31;
  int row0 = blockIdx.x * 16;
  if (row0 + 16 > n) return;
  int col0 = wave * 16;
  const _Float16* pa0 = h1  + (size_t)(row0 + (lane & 15)) * HC1 + ((lane >> 4) << 3);
  const _Float16* pb0 = w2t + (size_t)(col0 + (lane & 15)) * HC1 + ((lane >> 4) << 3);
  v8f c = {};
#pragma unroll
  for (int k = 0; k < HC1; k += 32) {
    v16h a = frag_load(pa0 + k);
    v16h b = frag_load(pb0 + k);
    c = __builtin_amdgcn_wmma_f32_16x16x32_f16(false, a, false, b, (short)0, c, false, false);
  }
  int rb = row0 + ((lane >> 4) << 3);
  int col = col0 + (lane & 15);
#pragma unroll
  for (int j = 0; j < 8; ++j) xh2[(size_t)(rb + j) * HC2 + col] = c[j];
}

// ---------- attention ----------
__global__ void k_attn_sd(const float* __restrict__ xh, int HC, int Ch,
                          const float* __restrict__ asr, const float* __restrict__ adr,
                          float* __restrict__ as_, float* __restrict__ ad_, int n) {
  int idx = blockIdx.x * blockDim.x + threadIdx.x;   // n*4 + h
  if (idx >= n * HH) return;
  int nn = idx >> 2, h = idx & 3;
  const float* xp = xh + (size_t)nn * HC + h * Ch;
  const float* s = asr + h * Ch; const float* d = adr + h * Ch;
  float vs = 0.f, vd = 0.f;
  for (int c = 0; c < Ch; ++c) { float v = xp[c]; vs += v * s[c]; vd += v * d[c]; }
  as_[idx] = vs; ad_[idx] = vd;
}
__global__ void k_logit_edges(const int* __restrict__ src, const int* __restrict__ dst,
                              const float* __restrict__ ea, const float* __restrict__ ve,
                              const float* __restrict__ as_, const float* __restrict__ ad_,
                              float* __restrict__ logit, unsigned* __restrict__ mflip, int e) {
  int i = blockIdx.x * blockDim.x + threadIdx.x;
  if (i >= e) return;
  int s = src[i], d = dst[i];
  float e0 = ea[(size_t)i * 4 + 0], e1 = ea[(size_t)i * 4 + 1];
  float e2 = ea[(size_t)i * 4 + 2], e3 = ea[(size_t)i * 4 + 3];
#pragma unroll
  for (int h = 0; h < HH; ++h) {
    float ae = e0 * ve[h * 4 + 0] + e1 * ve[h * 4 + 1] + e2 * ve[h * 4 + 2] + e3 * ve[h * 4 + 3];
    float l = as_[s * 4 + h] + ad_[d * 4 + h] + ae;
    l = (l > 0.f) ? l : NEG * l;
    logit[(size_t)i * 4 + h] = l;
    atomicMax(&mflip[d * 4 + h], flipf(l));
  }
}
__global__ void k_logit_loops(const float* __restrict__ lea, const float* __restrict__ ve,
                              const float* __restrict__ as_, const float* __restrict__ ad_,
                              float* __restrict__ logit, unsigned* __restrict__ mflip,
                              int e, int n) {
  int i = blockIdx.x * blockDim.x + threadIdx.x;
  if (i >= n) return;
  float e0 = lea[i * 4 + 0], e1 = lea[i * 4 + 1], e2 = lea[i * 4 + 2], e3 = lea[i * 4 + 3];
#pragma unroll
  for (int h = 0; h < HH; ++h) {
    float ae = e0 * ve[h * 4 + 0] + e1 * ve[h * 4 + 1] + e2 * ve[h * 4 + 2] + e3 * ve[h * 4 + 3];
    float l = as_[i * 4 + h] + ad_[i * 4 + h] + ae;
    l = (l > 0.f) ? l : NEG * l;
    logit[(size_t)(e + i) * 4 + h] = l;
    atomicMax(&mflip[i * 4 + h], flipf(l));
  }
}
__global__ void k_exden(const int* __restrict__ dst, const unsigned* __restrict__ mflip,
                        float* __restrict__ lg, float* __restrict__ den, int e, int n) {
  int k = blockIdx.x * blockDim.x + threadIdx.x;
  if (k >= e + n) return;
  int d = (k < e) ? dst[k] : (k - e);
#pragma unroll
  for (int h = 0; h < HH; ++h) {
    float m = unflipf(mflip[d * 4 + h]);
    float ex = __expf(lg[(size_t)k * 4 + h] - m);
    lg[(size_t)k * 4 + h] = ex;
    atomicAdd(&den[d * 4 + h], ex);
  }
}
// one wave per (edge|loop); lanes stripe channels
__global__ void k_agg(const int* __restrict__ src, const int* __restrict__ dst,
                      const float* __restrict__ ex, const float* __restrict__ den,
                      const float* __restrict__ xh, float* __restrict__ out,
                      int HC, int shift, int e, int n) {
  int k = blockIdx.x * 8 + (threadIdx.x >> 5);
  if (k >= e + n) return;
  int lane = threadIdx.x & 31;
  int s = (k < e) ? src[k] : (k - e);
  int d = (k < e) ? dst[k] : (k - e);
  for (int c = lane; c < HC; c += 32) {
    int h = c >> shift;
    float alpha = ex[(size_t)k * 4 + h] / den[d * 4 + h];
    atomicAdd(&out[(size_t)d * HC + c], alpha * xh[(size_t)s * HC + c]);
  }
}
__global__ void k_fin1(const float* __restrict__ o1, const float* __restrict__ b1,
                       _Float16* __restrict__ h1, int n) {
  int idx = blockIdx.x * blockDim.x + threadIdx.x;
  if (idx >= n * HC1) return;
  float v = o1[idx] + b1[idx & 255];
  v = (v > 0.f) ? v : 0.f;
  h1[idx] = (_Float16)v;
}
__global__ void k_bias2(float* __restrict__ out, const float* __restrict__ b2, int n) {
  int idx = blockIdx.x * blockDim.x + threadIdx.x;
  if (idx >= n * HC2) return;
  out[idx] += b2[idx & 127];
}

// ---------- launch ----------
extern "C" void kernel_launch(void* const* d_in, const int* in_sizes, int n_in,
                              void* d_out, int out_size, void* d_ws, size_t ws_size,
                              hipStream_t stream) {
  const float* x   = (const float*)d_in[0];
  const int*   ei  = (const int*)  d_in[1];
  const float* ea  = (const float*)d_in[2];
  const float* W1  = (const float*)d_in[3];
  const float* We1 = (const float*)d_in[4];
  const float* as1 = (const float*)d_in[5];
  const float* ad1 = (const float*)d_in[6];
  const float* ae1 = (const float*)d_in[7];
  const float* b1  = (const float*)d_in[8];
  const float* W2  = (const float*)d_in[9];
  const float* We2 = (const float*)d_in[10];
  const float* as2 = (const float*)d_in[11];
  const float* ad2 = (const float*)d_in[12];
  const float* ae2 = (const float*)d_in[13];
  const float* b2  = (const float*)d_in[14];
  float* out = (float*)d_out;

  const int n = in_sizes[0] / IND;       // 50000
  const int e = in_sizes[1] / 2;         // 800000
  const int* src = ei;
  const int* dst = ei + e;

  size_t off = 0;
  auto alloc = [&](size_t bytes) {
    void* p = (char*)d_ws + off;
    off += (bytes + 255) & ~(size_t)255;
    return p;
  };
  _Float16* xpad = (_Float16*)alloc((size_t)n * 32 * 2);
  _Float16* w1t  = (_Float16*)alloc((size_t)HC1 * 32 * 2);
  _Float16* w2t  = (_Float16*)alloc((size_t)HC2 * HC1 * 2);
  float*    xh1  = (float*)   alloc((size_t)n * HC1 * 4);
  float*    o1   = (float*)   alloc((size_t)n * HC1 * 4);
  _Float16* h1   = (_Float16*)alloc((size_t)n * HC1 * 2);
  float*    xh2  = (float*)   alloc((size_t)n * HC2 * 4);
  float*    as_  = (float*)   alloc((size_t)n * 4 * 4);
  float*    ad_  = (float*)   alloc((size_t)n * 4 * 4);
  float*    cnt  = (float*)   alloc((size_t)n * 4);
  float*    lea  = (float*)   alloc((size_t)n * 4 * 4);
  float*    lg   = (float*)   alloc((size_t)(e + n) * 4 * 4);
  unsigned* mflip= (unsigned*)alloc((size_t)n * 4 * 4);
  float*    den  = (float*)   alloc((size_t)n * 4 * 4);
  float*    ve   = (float*)   alloc(32 * 4);

  const int B = 256;
  auto g = [](long long t, int b) { return (int)((t + b - 1) / b); };

  // zero accumulators
  hipMemsetAsync(cnt,  0, (size_t)n * 4, stream);
  hipMemsetAsync(lea,  0, (size_t)n * 16, stream);
  hipMemsetAsync(mflip,0, (size_t)n * 16, stream);
  hipMemsetAsync(den,  0, (size_t)n * 16, stream);
  hipMemsetAsync(o1,   0, (size_t)n * HC1 * 4, stream);
  hipMemsetAsync(out,  0, (size_t)n * HC2 * 4, stream);

  // precompute / convert
  k_cvt_x <<<g((long long)n * 32, B), B, 0, stream>>>(x, xpad, n);
  k_cvt_w1<<<g(HC1 * 32, B), B, 0, stream>>>(W1, w1t);
  k_cvt_w2<<<g(HC2 * HC1, B), B, 0, stream>>>(W2, w2t);
  k_ve    <<<1, 32, 0, stream>>>(We1, ae1, We2, ae2, ve);
  k_cntsum<<<g(e, B), B, 0, stream>>>(dst, ea, cnt, lea, e);
  k_loopdiv<<<g(n * 4, B), B, 0, stream>>>(lea, cnt, n);

  // ---- layer 1 ----
  k_gemm1<<<n / 16, 256, 0, stream>>>(xpad, w1t, xh1, n);
  k_attn_sd<<<g(n * 4, B), B, 0, stream>>>(xh1, HC1, CC1, as1, ad1, as_, ad_, n);
  k_logit_edges<<<g(e, B), B, 0, stream>>>(src, dst, ea, ve, as_, ad_, lg, mflip, e);
  k_logit_loops<<<g(n, B), B, 0, stream>>>(lea, ve, as_, ad_, lg, mflip, e, n);
  k_exden<<<g(e + n, B), B, 0, stream>>>(dst, mflip, lg, den, e, n);
  k_agg<<<g(e + n, 8), 256, 0, stream>>>(src, dst, lg, den, xh1, o1, HC1, 6, e, n);
  k_fin1<<<g((long long)n * HC1, B), B, 0, stream>>>(o1, b1, h1, n);

  // ---- layer 2 ----
  hipMemsetAsync(mflip, 0, (size_t)n * 16, stream);
  hipMemsetAsync(den,   0, (size_t)n * 16, stream);
  k_gemm2<<<n / 16, 256, 0, stream>>>(h1, w2t, xh2, n);
  k_attn_sd<<<g(n * 4, B), B, 0, stream>>>(xh2, HC2, CC2, as2, ad2, as_, ad_, n);
  k_logit_edges<<<g(e, B), B, 0, stream>>>(src, dst, ea, ve + 16, as_, ad_, lg, mflip, e);
  k_logit_loops<<<g(n, B), B, 0, stream>>>(lea, ve + 16, as_, ad_, lg, mflip, e, n);
  k_exden<<<g(e + n, B), B, 0, stream>>>(dst, mflip, lg, den, e, n);
  k_agg<<<g(e + n, 8), 256, 0, stream>>>(src, dst, lg, den, xh2, out, HC2, 5, e, n);
  k_bias2<<<g((long long)n * HC2, B), B, 0, stream>>>(out, b2, n);
}
